// QuantDiT_14637248545469
// MI455X (gfx1250) — 16-bit and 8-bit matrix operand order repaired and checked by reading, compiled, not run on hardware
//
#include <hip/hip_runtime.h>
#include <math.h>


constexpr int HDIM = 1152, NHEAD = 16, HEADD = 72, IDIM = 4608;
constexpr int NLAYERS = 4, ODIM = 32, TD = 256;
constexpr int NB = 2, NQ = 1024, NKV = 300;
constexpr int MTOK = NB * NQ;
constexpr int ETOK = NB * NKV;
constexpr int ETOKP = 608;
constexpr int KT_H = HDIM / 64;
constexpr int KT_I = IDIM / 64;
constexpr int NT_H = HDIM / 16;
constexpr int NT_I = IDIM / 16;
constexpr int MT_X = MTOK / 16;
constexpr int MT_E = ETOKP / 16;
constexpr int QT16 = NQ / 16;
constexpr int DKT = 3;
constexpr int DNT = 5;

typedef __attribute__((ext_vector_type(8)))  int      v8i;
typedef __attribute__((ext_vector_type(8)))  float    v8f;
typedef __attribute__((ext_vector_type(16))) _Float16 v16h;

__device__ __forceinline__ float siluf(float x) { return x / (1.f + __expf(-x)); }
__device__ __forceinline__ float geluf(float x) { return 0.5f * x * (1.f + erff(x * 0.7071067811865475f)); }

__device__ __forceinline__ float wave_sum(float v) {
  for (int o = 16; o > 0; o >>= 1) v += __shfl_xor(v, o, 32);
  return v;
}
__device__ __forceinline__ float wave_max(float v) {
  for (int o = 16; o > 0; o >>= 1) v = fmaxf(v, __shfl_xor(v, o, 32));
  return v;
}
__device__ __forceinline__ float half_max(float v) {
  for (int m = 1; m < 16; m <<= 1) v = fmaxf(v, __shfl_xor(v, m, 32));
  return v;
}
__device__ __forceinline__ float half_sum(float v) {
  for (int m = 1; m < 16; m <<= 1) v += __shfl_xor(v, m, 32);
  return v;
}

__device__ __forceinline__ int a_frag_off(int mtile, int KT, int kt, int row16, int kk) {
  int h = (kk >> 3) & 1;
  int v = ((kk >> 5) << 2) + (((kk >> 4) & 1) << 1) + ((kk >> 2) & 1);
  int lane = row16 + (h << 4);
  return ((mtile * KT + kt) << 10) + lane * 32 + (v << 2);
}

__global__ void temb_kernel(const int* __restrict__ ts,
                            const float* __restrict__ w1, const float* __restrict__ b1,
                            const float* __restrict__ w2, const float* __restrict__ b2,
                            float* __restrict__ temb) {
  int b = blockIdx.x, t = threadIdx.x;
  __shared__ float e[TD];
  __shared__ float h1[HDIM];
  float tv = (float)ts[b];
  if (t < TD) {
    int half = TD / 2;
    int idx = (t < half) ? t : (t - half);
    float f = __expf(-logf(10000.f) * (float)idx / (float)half);
    float a = tv * f;
    e[t] = (t < half) ? sinf(a) : cosf(a);
  }
  __syncthreads();
  for (int j = t; j < HDIM; j += 256) {
    float acc = b1[j];
    for (int k = 0; k < TD; k++) acc += e[k] * w1[k * HDIM + j];
    h1[j] = siluf(acc);
  }
  __syncthreads();
  for (int j = t; j < HDIM; j += 256) {
    float acc = b2[j];
    for (int k = 0; k < HDIM; k++) acc += h1[k] * w2[k * HDIM + j];
    temb[b * HDIM + j] = acc;
  }
}

__global__ void silu_mlp_kernel(const float* __restrict__ temb, const float* __restrict__ W,
                                const float* __restrict__ bias, float* __restrict__ out, int Ncol) {
  int blocksPerB = Ncol >> 8;
  int b = blockIdx.x / blocksPerB;
  int j = (blockIdx.x % blocksPerB) * 256 + threadIdx.x;
  __shared__ float s[HDIM];
  for (int k = threadIdx.x; k < HDIM; k += 256) s[k] = siluf(temb[b * HDIM + k]);
  __syncthreads();
  float acc = bias[j];
  for (int k = 0; k < HDIM; k++) acc += s[k] * W[(size_t)k * Ncol + j];
  out[b * Ncol + j] = acc;
}

__global__ void wscale_kernel(const float* __restrict__ W, float* __restrict__ scale,
                              int K, int N) {
  int n = blockIdx.x * 256 + threadIdx.x;
  if (n >= N) return;
  float mx = 0.f;
  for (int k = 0; k < K; k++) mx = fmaxf(mx, fabsf(W[(size_t)k * N + n]));
  scale[n] = fmaxf(mx, 1e-8f) * (1.f / 127.f);
}

__global__ void wpack_kernel(const float* __restrict__ W, const float* __restrict__ scale,
                             signed char* __restrict__ frag, int KTiles, int NTiles, int N) {
  int wg = blockIdx.x * 8 + (threadIdx.x >> 5);
  int lane = threadIdx.x & 31;
  if (wg >= NTiles * KTiles) return;
  int nt = wg / KTiles, kt = wg % KTiles;
  int n = nt * 16 + (lane & 15);
  int half = lane >> 4;
  float inv = 1.f / scale[n];
  int base = ((nt * KTiles + kt) << 10) + lane * 32;
  for (int v = 0; v < 8; v++) {
    int kb = kt * 64 + ((v >> 2) << 5) + (((v >> 1) & 1) << 4) + (half << 3) + ((v & 1) << 2);
    unsigned int d = 0;
    for (int j = 0; j < 4; j++) {
      int q = (int)rintf(W[(size_t)(kb + j) * N + n] * inv);
      d |= ((unsigned int)(unsigned char)(signed char)q) << (8 * j);
    }
    *(unsigned int*)(frag + base + (v << 2)) = d;
  }
}

__global__ void __launch_bounds__(256)
ln_quant_kernel(const float* __restrict__ X, const float* __restrict__ emb,
                float* __restrict__ sOut, signed char* __restrict__ frag,
                int tokPerBatch, float eps) {
  int tok = blockIdx.x * 8 + (threadIdx.x >> 5);
  int lane = threadIdx.x & 31;
  int b = tok / tokPerBatch;
  const float* xr = X + (size_t)tok * HDIM;
  float r[36];
  float sum = 0.f;
#pragma unroll
  for (int t = 0; t < 9; t++) {
    float4 v = *(const float4*)(xr + (lane + 32 * t) * 4);
    r[4 * t + 0] = v.x; r[4 * t + 1] = v.y; r[4 * t + 2] = v.z; r[4 * t + 3] = v.w;
    sum += v.x + v.y + v.z + v.w;
  }
  float mean = wave_sum(sum) * (1.f / (float)HDIM);
  float vs = 0.f;
#pragma unroll
  for (int i = 0; i < 36; i++) { float d = r[i] - mean; vs += d * d; }
  float var = wave_sum(vs) * (1.f / (float)HDIM);
  float rstd = rsqrtf(var + eps);
  float amax = 0.f;
#pragma unroll
  for (int t = 0; t < 9; t++) {
#pragma unroll
    for (int j = 0; j < 4; j++) {
      int k = (lane + 32 * t) * 4 + j;
      float v = (r[4 * t + j] - mean) * rstd;
      if (emb) v = v * (1.f + emb[b * 2 * HDIM + k]) + emb[b * 2 * HDIM + HDIM + k];
      r[4 * t + j] = v;
      amax = fmaxf(amax, fabsf(v));
    }
  }
  amax = wave_max(amax);
  float s = fmaxf(amax, 1e-6f) * (1.f / 127.f);
  float inv = 1.f / s;
  if (lane == 0) sOut[tok] = s;
  int mt = tok >> 4, row = tok & 15;
#pragma unroll
  for (int t = 0; t < 9; t++) {
    int kg = (lane + 32 * t) * 4;
    unsigned int d = 0;
#pragma unroll
    for (int j = 0; j < 4; j++) {
      int q = (int)rintf(r[4 * t + j] * inv);
      d |= ((unsigned int)(unsigned char)(signed char)q) << (8 * j);
    }
    *(unsigned int*)(frag + a_frag_off(mt, KT_H, kg >> 6, row, kg & 63)) = d;
  }
}

__global__ void __launch_bounds__(256)
act_quant_kernel(const float* __restrict__ X, float* __restrict__ sOut,
                 signed char* __restrict__ frag, int K, int KT, int doGelu, int ntok) {
  int tok = blockIdx.x * 8 + (threadIdx.x >> 5);
  if (tok >= ntok) return;
  int lane = threadIdx.x & 31;
  const float* xr = X + (size_t)tok * K;
  float amax = 0.f;
  for (int k0 = lane * 4; k0 < K; k0 += 128) {
    float4 v = *(const float4*)(xr + k0);
    float a = doGelu ? geluf(v.x) : v.x;
    float b = doGelu ? geluf(v.y) : v.y;
    float c = doGelu ? geluf(v.z) : v.z;
    float d = doGelu ? geluf(v.w) : v.w;
    amax = fmaxf(amax, fmaxf(fmaxf(fabsf(a), fabsf(b)), fmaxf(fabsf(c), fabsf(d))));
  }
  amax = wave_max(amax);
  float s = fmaxf(amax, 1e-6f) * (1.f / 127.f);
  float inv = 1.f / s;
  if (lane == 0) sOut[tok] = s;
  int mt = tok >> 4, row = tok & 15;
  for (int k0 = lane * 4; k0 < K; k0 += 128) {
    float4 v = *(const float4*)(xr + k0);
    float vv[4] = {v.x, v.y, v.z, v.w};
    unsigned int d = 0;
    for (int j = 0; j < 4; j++) {
      float f = doGelu ? geluf(vv[j]) : vv[j];
      int q = (int)rintf(f * inv);
      d |= ((unsigned int)(unsigned char)(signed char)q) << (8 * j);
    }
    *(unsigned int*)(frag + a_frag_off(mt, KT, k0 >> 6, row, k0 & 63)) = d;
  }
}

__global__ void __launch_bounds__(256)
gemm_w8a8_kernel(const signed char* __restrict__ Af, const signed char* __restrict__ Bf,
                 const float* __restrict__ sx, const float* __restrict__ wsc,
                 const float* __restrict__ bias, const float* __restrict__ resid,
                 float* __restrict__ out, int MT, int NT, int KT, int N) {
  int w = threadIdx.x >> 5, lane = threadIdx.x & 31;
  int nt = blockIdx.x;
  int mt = blockIdx.y * 8 + w;
  if (mt >= MT) return;
  v8i acc = {0, 0, 0, 0, 0, 0, 0, 0};
  const signed char* ap = Af + (((size_t)mt * KT) << 10) + lane * 32;
  const signed char* bp = Bf + (((size_t)nt * KT) << 10) + lane * 32;
  for (int kt = 0; kt < KT; kt++) {
    v8i a = *(const v8i*)ap; ap += 1024;
    v8i b = *(const v8i*)bp; bp += 1024;
    acc = __builtin_amdgcn_wmma_i32_16x16x64_iu8(true, a, true, b, acc, false, false);
  }
  int col = nt * 16 + (lane & 15);
  float wv = wsc[col];
  float bv = bias[col];
  int rbase = mt * 16 + ((lane >> 4) << 3);
#pragma unroll
  for (int i = 0; i < 8; i++) {
    int row = rbase + i;
    float v = (float)acc[i] * sx[row] * wv + bv;
    size_t o = (size_t)row * N + col;
    if (resid) v += resid[o];
    out[o] = v;
  }
}

__global__ void __launch_bounds__(256)
qpack_kernel(const float* __restrict__ qbuf, char* __restrict__ frag) {
  int wg = blockIdx.x * 8 + (threadIdx.x >> 5);
  int lane = threadIdx.x & 31;
  int dkt = wg % DKT;
  int qt = (wg / DKT) % QT16;
  int h = (wg / (DKT * QT16)) % NHEAD;
  int b = wg / (DKT * QT16 * NHEAD);
  int M = lane & 15, hf = lane >> 4;
  const float* src = qbuf + ((size_t)(b * NQ + qt * 16 + M)) * HDIM + h * HEADD;
  v16h t;
#pragma unroll
  for (int v = 0; v < 8; v++) {
    int k0 = hf * 16 + 2 * v;
    int d0 = dkt * 32 + k0;
    t[2 * v] = (_Float16)((d0 < HEADD) ? src[d0] : 0.f);
    t[2 * v + 1] = (_Float16)((d0 + 1 < HEADD) ? src[d0 + 1] : 0.f);
  }
  *(v16h*)(frag + (((size_t)wg) << 10) + lane * 32) = t;
}

__global__ void __launch_bounds__(256)
kpack_kernel(const float* __restrict__ kbuf, char* __restrict__ frag,
             int KVT16, int KV, int kvStride) {
  int wg = blockIdx.x * 8 + (threadIdx.x >> 5);
  int lane = threadIdx.x & 31;
  int dkt = wg % DKT;
  int kvt = (wg / DKT) % KVT16;
  int h = (wg / (DKT * KVT16)) % NHEAD;
  int b = wg / (DKT * KVT16 * NHEAD);
  int n = lane & 15, hf = lane >> 4;
  int col = kvt * 16 + n;
  bool cv = col < KV;
  const float* src = kbuf + ((size_t)(b * kvStride + (cv ? col : 0))) * HDIM + h * HEADD;
  v16h t;
#pragma unroll
  for (int v = 0; v < 8; v++) {
    int d0 = dkt * 32 + hf * 16 + 2 * v;
    t[2 * v] = (_Float16)((cv && d0 < HEADD) ? src[d0] : 0.f);
    t[2 * v + 1] = (_Float16)((cv && d0 + 1 < HEADD) ? src[d0 + 1] : 0.f);
  }
  *(v16h*)(frag + (((size_t)wg) << 10) + lane * 32) = t;
}

__global__ void __launch_bounds__(256)
vpack_kernel(const float* __restrict__ vbuf, char* __restrict__ frag,
             int KVT32, int KV, int kvStride) {
  int wg = blockIdx.x * 8 + (threadIdx.x >> 5);
  int lane = threadIdx.x & 31;
  int dnt = wg % DNT;
  int kvt = (wg / DNT) % KVT32;
  int h = (wg / (DNT * KVT32)) % NHEAD;
  int b = wg / (DNT * KVT32 * NHEAD);
  int n = lane & 15, hf = lane >> 4;
  int dcol = dnt * 16 + n;
  bool dv = dcol < HEADD;
  v16h t;
#pragma unroll
  for (int v = 0; v < 8; v++) {
    int kv0 = kvt * 32 + hf * 16 + 2 * v;
    float f0 = (dv && kv0 < KV) ? vbuf[((size_t)(b * kvStride + kv0)) * HDIM + h * HEADD + dcol] : 0.f;
    float f1 = (dv && kv0 + 1 < KV) ? vbuf[((size_t)(b * kvStride + kv0 + 1)) * HDIM + h * HEADD + dcol] : 0.f;
    t[2 * v] = (_Float16)f0;
    t[2 * v + 1] = (_Float16)f1;
  }
  *(v16h*)(frag + (((size_t)wg) << 10) + lane * 32) = t;
}

__global__ void __launch_bounds__(256)
fattn_kernel(const char* __restrict__ qf, const char* __restrict__ kf,
             const char* __restrict__ vf, float* __restrict__ ao,
             int KV, int KVC32, int KVT16) {
  __shared__ float pt[8][16][36];
  int w = threadIdx.x >> 5, lane = threadIdx.x & 31;
  int wg = blockIdx.x * 8 + w;
  int qt = wg % QT16;
  int h = (wg / QT16) % NHEAD;
  int b = wg / (QT16 * NHEAD);
  int lcol = lane & 15, lhalf = lane >> 4;
  const float scl = 0.11785113019775793f;

  size_t qbase = (((size_t)wg) * DKT) << 10;
  v16h q0 = *(const v16h*)(qf + qbase + 0 * 1024 + lane * 32);
  v16h q1 = *(const v16h*)(qf + qbase + 1 * 1024 + lane * 32);
  v16h q2 = *(const v16h*)(qf + qbase + 2 * 1024 + lane * 32);

  const v8f zf = {0.f, 0.f, 0.f, 0.f, 0.f, 0.f, 0.f, 0.f};
  float mrow[8], srow[8];
  v8f o[DNT];
#pragma unroll
  for (int i = 0; i < 8; i++) { mrow[i] = -3.0e30f; srow[i] = 0.f; }
#pragma unroll
  for (int d = 0; d < DNT; d++) o[d] = zf;

  for (int ch = 0; ch < KVC32; ch++) {
    const char* kb0 = kf + ((((size_t)((b * NHEAD + h) * KVT16 + ch * 2)) * DKT) << 10) + lane * 32;
    const char* kb1 = kb0 + (size_t)DKT * 1024;
    v8f s0 = zf, s1 = zf;
    s0 = __builtin_amdgcn_wmma_f32_16x16x32_f16(false, q0, false, *(const v16h*)(kb0 + 0 * 1024), (short)0, s0, false, false);
    s0 = __builtin_amdgcn_wmma_f32_16x16x32_f16(false, q1, false, *(const v16h*)(kb0 + 1 * 1024), (short)0, s0, false, false);
    s0 = __builtin_amdgcn_wmma_f32_16x16x32_f16(false, q2, false, *(const v16h*)(kb0 + 2 * 1024), (short)0, s0, false, false);
    s1 = __builtin_amdgcn_wmma_f32_16x16x32_f16(false, q0, false, *(const v16h*)(kb1 + 0 * 1024), (short)0, s1, false, false);
    s1 = __builtin_amdgcn_wmma_f32_16x16x32_f16(false, q1, false, *(const v16h*)(kb1 + 1 * 1024), (short)0, s1, false, false);
    s1 = __builtin_amdgcn_wmma_f32_16x16x32_f16(false, q2, false, *(const v16h*)(kb1 + 2 * 1024), (short)0, s1, false, false);

    int col0 = ch * 32 + lcol;
    bool v0 = col0 < KV, v1 = col0 + 16 < KV;
#pragma unroll
    for (int i = 0; i < 8; i++) {
      float a0 = v0 ? s0[i] * scl : -3.0e30f;
      float a1 = v1 ? s1[i] * scl : -3.0e30f;
      float cm = half_max(fmaxf(a0, a1));
      float mnew = fmaxf(mrow[i], cm);
      float corr = __expf(mrow[i] - mnew);
      float p0 = v0 ? __expf(a0 - mnew) : 0.f;
      float p1 = v1 ? __expf(a1 - mnew) : 0.f;
      float cs = half_sum(p0 + p1);
      srow[i] = srow[i] * corr + cs;
      mrow[i] = mnew;
#pragma unroll
      for (int d = 0; d < DNT; d++) o[d][i] *= corr;
      pt[w][i + 8 * lhalf][lcol] = p0;
      pt[w][i + 8 * lhalf][16 + lcol] = p1;
    }
    asm volatile("" ::: "memory");
    v16h pa;
#pragma unroll
    for (int v = 0; v < 8; v++) {
      int k0 = lhalf * 16 + 2 * v;
      float2 f = *(const float2*)(&pt[w][lcol][k0]);
      pa[2 * v] = (_Float16)f.x;
      pa[2 * v + 1] = (_Float16)f.y;
    }
    asm volatile("" ::: "memory");
    const char* vb0 = vf + ((((size_t)((b * NHEAD + h) * KVC32 + ch)) * DNT) << 10) + lane * 32;
#pragma unroll
    for (int d = 0; d < DNT; d++)
      o[d] = __builtin_amdgcn_wmma_f32_16x16x32_f16(false, pa, false, *(const v16h*)(vb0 + (size_t)d * 1024), (short)0, o[d], false, false);
  }

#pragma unroll
  for (int d = 0; d < DNT; d++) {
    int dcol = d * 16 + lcol;
    if (dcol < HEADD) {
#pragma unroll
      for (int i = 0; i < 8; i++) {
        int qrow = qt * 16 + i + 8 * lhalf;
        ao[((size_t)(b * NQ + qrow)) * HDIM + h * HEADD + dcol] = o[d][i] / srow[i];
      }
    }
  }
}

__global__ void __launch_bounds__(256)
head_kernel(const float* __restrict__ hbuf, const float* __restrict__ cond,
            const float* __restrict__ W, const float* __restrict__ bias,
            float* __restrict__ out) {
  __shared__ float nh[8][HDIM];
  int w = threadIdx.x >> 5, lane = threadIdx.x & 31;
  int tok = blockIdx.x * 8 + w;
  int b = tok / NQ;
  const float* xr = hbuf + (size_t)tok * HDIM;
  float r[36];
  float sum = 0.f;
#pragma unroll
  for (int t = 0; t < 9; t++) {
    float4 v = *(const float4*)(xr + (lane + 32 * t) * 4);
    r[4 * t] = v.x; r[4 * t + 1] = v.y; r[4 * t + 2] = v.z; r[4 * t + 3] = v.w;
    sum += v.x + v.y + v.z + v.w;
  }
  float mean = wave_sum(sum) * (1.f / (float)HDIM);
  float vs = 0.f;
#pragma unroll
  for (int i = 0; i < 36; i++) { float d = r[i] - mean; vs += d * d; }
  float rstd = rsqrtf(wave_sum(vs) * (1.f / (float)HDIM) + 1e-6f);
#pragma unroll
  for (int t = 0; t < 9; t++) {
#pragma unroll
    for (int j = 0; j < 4; j++) {
      int k = (lane + 32 * t) * 4 + j;
      float shift = cond[b * 2 * HDIM + k];
      float scale = cond[b * 2 * HDIM + HDIM + k];
      nh[w][k] = (r[4 * t + j] - mean) * rstd * (1.f + scale) + shift;
    }
  }
  __syncthreads();
  float acc = bias[lane];
  for (int k = 0; k < HDIM; k++) acc += nh[w][k] * W[k * ODIM + lane];
  out[(size_t)tok * ODIM + lane] = acc;
}

extern "C" void kernel_launch(void* const* d_in, const int* in_sizes, int n_in,
                              void* d_out, int out_size, void* d_ws, size_t ws_size,
                              hipStream_t stream) {
  (void)in_sizes; (void)n_in; (void)out_size; (void)ws_size;
  const float* hs   = (const float*)d_in[0];
  const float* ehs  = (const float*)d_in[1];
  const int*   tsp  = (const int*)d_in[2];
  const float* t_w1 = (const float*)d_in[3];
  const float* t_b1 = (const float*)d_in[4];
  const float* t_w2 = (const float*)d_in[5];
  const float* t_b2 = (const float*)d_in[6];
  const float* po1w = (const float*)d_in[7];
  const float* po1b = (const float*)d_in[8];
  const float* po2w = (const float*)d_in[9];
  const float* po2b = (const float*)d_in[10];
  auto LP = [&](int l, int o) { return (const float*)d_in[11 + l * 14 + o]; };

  char* cur = (char*)d_ws;
  auto alloc = [&](size_t b) -> char* {
    char* r = cur; cur += (b + 255) & ~(size_t)255; return r;
  };
  float* temb = (float*)alloc(NB * HDIM * 4);
  float* cond = (float*)alloc(NB * 2 * HDIM * 4);
  float* emb  = (float*)alloc(NB * 2 * HDIM * 4);
  float* se   = (float*)alloc(ETOKP * 4);
  float* sx   = (float*)alloc(MTOK * 4);
  float* sa   = (float*)alloc(MTOK * 4);
  float* sg   = (float*)alloc(MTOK * 4);
  float* wsq  = (float*)alloc(HDIM * 4);
  float* wsk  = (float*)alloc(HDIM * 4);
  float* wsv  = (float*)alloc(HDIM * 4);
  float* wso  = (float*)alloc(HDIM * 4);
  float* wsf1 = (float*)alloc(IDIM * 4);
  float* wsf2 = (float*)alloc(HDIM * 4);
  signed char* qeF = (signed char*)alloc((size_t)MT_E * KT_H * 1024);
  signed char* qxF = (signed char*)alloc((size_t)MT_X * KT_H * 1024);
  signed char* qaF = (signed char*)alloc((size_t)MT_X * KT_H * 1024);
  signed char* qgF = (signed char*)alloc((size_t)MT_X * KT_I * 1024);
  signed char* wqF[4];
  for (int j = 0; j < 4; j++) wqF[j] = (signed char*)alloc((size_t)NT_H * KT_H * 1024);
  signed char* wf1F = (signed char*)alloc((size_t)NT_I * KT_H * 1024);
  signed char* wf2F = (signed char*)alloc((size_t)NT_H * KT_I * 1024);
  char* qfB = alloc((size_t)NB * NHEAD * QT16 * DKT * 1024);
  char* kfB = alloc((size_t)NB * NHEAD * QT16 * DKT * 1024);
  char* vfB = alloc((size_t)NB * NHEAD * 32 * DNT * 1024);
  float* hbuf = (float*)alloc((size_t)MTOK * HDIM * 4);
  float* ao   = (float*)alloc((size_t)MTOK * HDIM * 4);
  char*  big  = alloc((size_t)MTOK * IDIM * 4);
  float* qb  = (float*)big;
  float* kb  = (float*)(big + (size_t)MTOK * HDIM * 4);
  float* vb  = (float*)(big + 2 * (size_t)MTOK * HDIM * 4);
  float* f1o = (float*)big;

  temb_kernel<<<NB, 256, 0, stream>>>(tsp, t_w1, t_b1, t_w2, t_b2, temb);
  silu_mlp_kernel<<<NB * 2 * HDIM / 256, 256, 0, stream>>>(temb, po1w, po1b, cond, 2 * HDIM);
  hipMemcpyAsync(hbuf, hs, (size_t)MTOK * HDIM * 4, hipMemcpyDeviceToDevice, stream);
  act_quant_kernel<<<ETOK / 8, 256, 0, stream>>>(ehs, se, qeF, HDIM, KT_H, 0, ETOK);

  for (int l = 0; l < NLAYERS; l++) {
    const float* adaw = LP(l, 0);  const float* adab = LP(l, 1);
    const float* qw   = LP(l, 2);  const float* qbi  = LP(l, 3);
    const float* kw   = LP(l, 4);  const float* kbi  = LP(l, 5);
    const float* vw   = LP(l, 6);  const float* vbi  = LP(l, 7);
    const float* ow   = LP(l, 8);  const float* obi  = LP(l, 9);
    const float* f1w  = LP(l, 10); const float* f1b  = LP(l, 11);
    const float* f2w  = LP(l, 12); const float* f2b  = LP(l, 13);
    bool cross = (l % 2 == 0);
    int KV = cross ? NKV : NQ;
    int KVT16 = cross ? 20 : 64;
    int KVC32 = cross ? 10 : 32;
    int kvStride = cross ? NKV : NQ;

    silu_mlp_kernel<<<NB * 2 * HDIM / 256, 256, 0, stream>>>(temb, adaw, adab, emb, 2 * HDIM);

    const float* Ws[6] = {qw, kw, vw, ow, f1w, f2w};
    float* Ss[6] = {wsq, wsk, wsv, wso, wsf1, wsf2};
    signed char* Fs[6] = {wqF[0], wqF[1], wqF[2], wqF[3], wf1F, wf2F};
    int Ks[6] = {HDIM, HDIM, HDIM, HDIM, HDIM, IDIM};
    int Ns[6] = {HDIM, HDIM, HDIM, HDIM, IDIM, HDIM};
    for (int j = 0; j < 6; j++) {
      int K = Ks[j], N = Ns[j], kt = K / 64, ntt = N / 16;
      wscale_kernel<<<(N + 255) / 256, 256, 0, stream>>>(Ws[j], Ss[j], K, N);
      wpack_kernel<<<(ntt * kt + 7) / 8, 256, 0, stream>>>(Ws[j], Ss[j], Fs[j], kt, ntt, N);
    }

    ln_quant_kernel<<<MTOK / 8, 256, 0, stream>>>(hbuf, emb, sx, qxF, NQ, 1e-5f);
    gemm_w8a8_kernel<<<dim3(NT_H, MT_X / 8), 256, 0, stream>>>(
        qxF, wqF[0], sx, wsq, qbi, nullptr, qb, MT_X, NT_H, KT_H, HDIM);
    if (cross) {
      gemm_w8a8_kernel<<<dim3(NT_H, (MT_E + 7) / 8), 256, 0, stream>>>(
          qeF, wqF[1], se, wsk, kbi, nullptr, kb, MT_E, NT_H, KT_H, HDIM);
      gemm_w8a8_kernel<<<dim3(NT_H, (MT_E + 7) / 8), 256, 0, stream>>>(
          qeF, wqF[2], se, wsv, vbi, nullptr, vb, MT_E, NT_H, KT_H, HDIM);
    } else {
      gemm_w8a8_kernel<<<dim3(NT_H, MT_X / 8), 256, 0, stream>>>(
          qxF, wqF[1], sx, wsk, kbi, nullptr, kb, MT_X, NT_H, KT_H, HDIM);
      gemm_w8a8_kernel<<<dim3(NT_H, MT_X / 8), 256, 0, stream>>>(
          qxF, wqF[2], sx, wsv, vbi, nullptr, vb, MT_X, NT_H, KT_H, HDIM);
    }
    qpack_kernel<<<NB * NHEAD * QT16 * DKT / 8, 256, 0, stream>>>(qb, qfB);
    kpack_kernel<<<NB * NHEAD * KVT16 * DKT / 8, 256, 0, stream>>>(kb, kfB, KVT16, KV, kvStride);
    vpack_kernel<<<NB * NHEAD * KVC32 * DNT / 8, 256, 0, stream>>>(vb, vfB, KVC32, KV, kvStride);
    fattn_kernel<<<NB * NHEAD * QT16 / 8, 256, 0, stream>>>(qfB, kfB, vfB, ao, KV, KVC32, KVT16);
    act_quant_kernel<<<MTOK / 8, 256, 0, stream>>>(ao, sa, qaF, HDIM, KT_H, 0, MTOK);
    gemm_w8a8_kernel<<<dim3(NT_H, MT_X / 8), 256, 0, stream>>>(
        qaF, wqF[3], sa, wso, obi, hbuf, hbuf, MT_X, NT_H, KT_H, HDIM);

    ln_quant_kernel<<<MTOK / 8, 256, 0, stream>>>(hbuf, nullptr, sx, qxF, NQ, 1e-5f);
    gemm_w8a8_kernel<<<dim3(NT_I, MT_X / 8), 256, 0, stream>>>(
        qxF, wf1F, sx, wsf1, f1b, nullptr, f1o, MT_X, NT_I, KT_H, IDIM);
    act_quant_kernel<<<MTOK / 8, 256, 0, stream>>>(f1o, sg, qgF, IDIM, KT_I, 1, MTOK);
    gemm_w8a8_kernel<<<dim3(NT_H, MT_X / 8), 256, 0, stream>>>(
        qgF, wf2F, sg, wsf2, f2b, hbuf, hbuf, MT_X, NT_H, KT_I, HDIM);
  }

  head_kernel<<<MTOK / 8, 256, 0, stream>>>(hbuf, cond, po2w, po2b, (float*)d_out);
}
